// mHCBlock_89721866813904
// MI455X (gfx1250) — matrix operand order repaired and checked by reading, not compiled, not run on hardware
//
#include <hip/hip_runtime.h>
#include <cstdint>
#include <cstddef>

// ---------------- CDNA5 (gfx1250, wave32) mHC block ----------------
// B=2, T=2048, D=2048, H=16, hd=128, streams=8, stream_dim=256.
// All GEMM-class math routed through v_wmma_f32_16x16x32_bf16.

typedef __attribute__((ext_vector_type(16))) __bf16 v16bf;
typedef __attribute__((ext_vector_type(8)))  __bf16 v8bf;
typedef __attribute__((ext_vector_type(8)))  float  v8f;

#define D_MODEL 2048
#define T_SEQ   2048
#define NHEADS  16
#define HDIM    128
#define NSTR    8
#define SDIM    256

__device__ inline v8f zero8() { v8f z = {0.f,0.f,0.f,0.f,0.f,0.f,0.f,0.f}; return z; }

__device__ inline v16bf make16(v8bf lo, v8bf hv) {
  return __builtin_shufflevector(lo, hv, 0,1,2,3,4,5,6,7,8,9,10,11,12,13,14,15);
}
// A-fragment (16x32 bf16, ISA 7.12.2): per-lane row already applied in p.
// lane-half hi selects K chunks [hi*8, hi*8+7] and [hi*8+16, hi*8+23].
__device__ inline v16bf load_a(const __bf16* p, int hi) {
  v8bf lo = *(const v8bf*)(p + hi * 8);
  v8bf hv = *(const v8bf*)(p + hi * 8 + 16);
  return make16(lo, hv);
}
// B-fragment (32x16 bf16): p points at this lane's column row (contiguous K);
// lane-half hi selects K chunks [hi*8, hi*8+7] and [hi*8+16, hi*8+23] (matches A).
__device__ inline v16bf load_b(const __bf16* p, int hi) {
  v8bf lo = *(const v8bf*)(p + hi * 8);
  v8bf hv = *(const v8bf*)(p + hi * 8 + 16);
  return make16(lo, hv);
}
__device__ inline v8f wmma_bf16(v16bf a, v16bf b, v8f c) {
  // (neg_a, A, neg_b, B, c_mod, C, reuse_a, reuse_b)
  return __builtin_amdgcn_wmma_f32_16x16x32_bf16(false, a, false, b, (short)0, c, false, false);
}

// ---------------- small prep kernels ----------------

__global__ void mean_kernel(const float* __restrict__ x, float* __restrict__ xs) {
  int idx = blockIdx.x * 256 + threadIdx.x;        // b*2048 + d
  int b = idx >> 11, d = idx & 2047;
  const float* p = x + (size_t)b * T_SEQ * D_MODEL + d;
  float s = 0.f;
  for (int t = 0; t < T_SEQ; ++t) s += p[(size_t)t * D_MODEL];
  xs[idx] = s * (1.0f / T_SEQ);
}

__global__ void gate_kernel(const float* __restrict__ xs, const float* __restrict__ gw,
                            const float* __restrict__ gb, float* __restrict__ g) {
  __shared__ float red[256];
  int b = blockIdx.x >> 3, o = blockIdx.x & 7;
  int j = 2 * NSTR + o;                             // residual gate rows
  float s = 0.f;
  for (int k = threadIdx.x; k < D_MODEL; k += 256)
    s += xs[b * D_MODEL + k] * gw[(size_t)j * D_MODEL + k];
  red[threadIdx.x] = s; __syncthreads();
  for (int st = 128; st > 0; st >>= 1) {
    if (threadIdx.x < st) red[threadIdx.x] += red[threadIdx.x + st];
    __syncthreads();
  }
  if (threadIdx.x == 0) {
    float z = red[0] + gb[j];
    g[b * NSTR + o] = 1.0f / (1.0f + __expf(-z));
  }
}

__global__ void sinkhorn_kernel(const float* __restrict__ phi, float* __restrict__ H) {
  __shared__ float Msh[64];
  int t = threadIdx.x;                              // 64 threads
  Msh[t] = __expf(phi[t]);
  __syncthreads();
  int r = t >> 3, c = t & 7;
  for (int it = 0; it < 15; ++it) {
    float rs = 0.f;
    for (int j = 0; j < 8; ++j) rs += Msh[r * 8 + j];
    float v = Msh[t] / rs;
    __syncthreads(); Msh[t] = v; __syncthreads();
    float cs = 0.f;
    for (int i = 0; i < 8; ++i) cs += Msh[i * 8 + c];
    v = Msh[t] / cs;
    __syncthreads(); Msh[t] = v; __syncthreads();
  }
  H[t] = Msh[t];
}

// out[b,t,m,d] = sum_n H[m,n] * g[b,n] * x[b,t,n,d]
__global__ void mhc_kernel(const float* __restrict__ x, const float* __restrict__ H,
                           const float* __restrict__ g, float* __restrict__ out) {
  __shared__ float Hs[64];
  __shared__ float gs[NSTR];
  const int bt = blockIdx.x;                        // b*T + t
  const int b = bt >> 11;
  if (threadIdx.x < 64) Hs[threadIdx.x] = H[threadIdx.x];
  if (threadIdx.x < NSTR) gs[threadIdx.x] = g[b * NSTR + threadIdx.x];
  __syncthreads();
  const float* xr = x + (size_t)bt * D_MODEL + threadIdx.x;
  float xv[NSTR];
  #pragma unroll
  for (int n = 0; n < NSTR; ++n) xv[n] = xr[n * SDIM] * gs[n];
  float* orow = out + (size_t)bt * D_MODEL + threadIdx.x;
  #pragma unroll
  for (int m = 0; m < NSTR; ++m) {
    float a = 0.f;
    #pragma unroll
    for (int n = 0; n < NSTR; ++n) a += Hs[m * 8 + n] * xv[n];
    orow[m * SDIM] = a;
  }
}

__global__ void rmsnorm_bf16_kernel(const float* __restrict__ x, const float* __restrict__ w,
                                    __bf16* __restrict__ out) {
  __shared__ float red[256];
  const size_t row = blockIdx.x;
  const float* xr = x + row * D_MODEL;
  float v[8]; float s = 0.f;
  #pragma unroll
  for (int i = 0; i < 8; ++i) { v[i] = xr[threadIdx.x + i * 256]; s += v[i] * v[i]; }
  red[threadIdx.x] = s; __syncthreads();
  for (int st = 128; st > 0; st >>= 1) {
    if (threadIdx.x < st) red[threadIdx.x] += red[threadIdx.x + st];
    __syncthreads();
  }
  float scale = rsqrtf(red[0] * (1.0f / D_MODEL) + 1e-6f);
  __bf16* orow = out + row * D_MODEL;
  #pragma unroll
  for (int i = 0; i < 8; ++i)
    orow[threadIdx.x + i * 256] = (__bf16)(v[i] * scale * w[threadIdx.x + i * 256]);
}

__global__ void f32_to_bf16_kernel(const float* __restrict__ in, __bf16* __restrict__ out) {
  size_t i = (size_t)blockIdx.x * 256 + threadIdx.x;
  out[i] = (__bf16)in[i];
}

// ---------------- WMMA GEMM: C[m,n] = sum_k A[m,k]*W[n,k] (+epilogue) ----------------
// MODE 0: QKV scatter -> Qh/Kh [B,H,T,hd] bf16 (Q pre-scaled by 1/sqrt(hd)), VhT [B,H,hd,T] bf16
// MODE 1: fout = C + bias + resid1              (out_proj -> x2)
// MODE 2: bout = bf16(gelu(C + bias))           (mlp1 -> h)
// MODE 3: fout = C + bias + resid1 + resid2     (mlp2 -> d_out, resid2 = mhc in d_out)
template <int MODE>
__global__ void __launch_bounds__(256)
gemm_wmma_kernel(const __bf16* __restrict__ A, const __bf16* __restrict__ W,
                 const float* __restrict__ bias, int M, int N, int K,
                 float* __restrict__ fout, __bf16* __restrict__ bout,
                 const float* __restrict__ resid1, const float* __restrict__ resid2,
                 __bf16* __restrict__ qbuf, __bf16* __restrict__ kbuf,
                 __bf16* __restrict__ vbuf) {
  __shared__ __bf16 ldsA[128 * 40];
  __shared__ __bf16 ldsW[128 * 40];
  const int tid = threadIdx.x;
  const int lane = tid & 31, wid = tid >> 5;
  const int hi = lane >> 4, ln = lane & 15;
  const int bm = blockIdx.y * 128, bn = blockIdx.x * 128;
  const int wM = (wid >> 2) * 64, wN = (wid & 3) * 32;   // 2x4 wave grid, 64x32 per wave

  v8f acc[4][2];
  #pragma unroll
  for (int i = 0; i < 4; ++i)
    #pragma unroll
    for (int j = 0; j < 2; ++j) acc[i][j] = zero8();

  const int lrow = tid >> 1, lhalf = tid & 1;            // 128 rows x 2 half-rows of 16 bf16
  for (int kt = 0; kt < K; kt += 32) {
    {
      const uint4* pa = (const uint4*)(A + (size_t)(bm + lrow) * K + kt + lhalf * 16);
      uint4 a0 = pa[0], a1 = pa[1];
      *(uint4*)&ldsA[lrow * 40 + lhalf * 16]     = a0;
      *(uint4*)&ldsA[lrow * 40 + lhalf * 16 + 8] = a1;
      const uint4* pw = (const uint4*)(W + (size_t)(bn + lrow) * K + kt + lhalf * 16);
      uint4 w0 = pw[0], w1 = pw[1];
      *(uint4*)&ldsW[lrow * 40 + lhalf * 16]     = w0;
      *(uint4*)&ldsW[lrow * 40 + lhalf * 16 + 8] = w1;
    }
    __syncthreads();
    v16bf af[4], bfv[2];
    #pragma unroll
    for (int i = 0; i < 4; ++i) af[i] = load_a(&ldsA[(wM + i * 16 + ln) * 40], hi);
    #pragma unroll
    for (int j = 0; j < 2; ++j) bfv[j] = load_b(&ldsW[(wN + j * 16 + ln) * 40], hi);
    #pragma unroll
    for (int i = 0; i < 4; ++i)
      #pragma unroll
      for (int j = 0; j < 2; ++j) acc[i][j] = wmma_bf16(af[i], bfv[j], acc[i][j]);
    __syncthreads();
  }

  #pragma unroll
  for (int i = 0; i < 4; ++i) {
    #pragma unroll
    for (int j = 0; j < 2; ++j) {
      #pragma unroll
      for (int r = 0; r < 8; ++r) {
        int m = bm + wM + i * 16 + hi * 8 + r;
        int n = bn + wN + j * 16 + ln;
        float cv = acc[i][j][r] + bias[n];
        if constexpr (MODE == 0) {
          int b = m >> 11, t = m & 2047;
          int which = n >> 11, h = (n & 2047) >> 7, d = n & 127;
          size_t bh = (size_t)(b * NHEADS + h);
          if (which == 0)
            qbuf[(bh * T_SEQ + t) * HDIM + d] = (__bf16)(cv * 0.08838834764831845f);
          else if (which == 1)
            kbuf[(bh * T_SEQ + t) * HDIM + d] = (__bf16)cv;
          else
            vbuf[(bh * HDIM + d) * T_SEQ + t] = (__bf16)cv;
        } else if constexpr (MODE == 1) {
          size_t idx = (size_t)m * N + n;
          fout[idx] = cv + resid1[idx];
        } else if constexpr (MODE == 2) {
          size_t idx = (size_t)m * N + n;
          bout[idx] = (__bf16)(0.5f * cv * (1.0f + erff(cv * 0.7071067811865475f)));
        } else {
          size_t idx = (size_t)m * N + n;
          fout[idx] = cv + resid1[idx] + resid2[idx];
        }
      }
    }
  }
}

// ---------------- flash attention (causal, wave-per-16-q-rows) ----------------
__global__ void __launch_bounds__(256)
attention_kernel(const __bf16* __restrict__ Qh, const __bf16* __restrict__ Kh,
                 const __bf16* __restrict__ VhT, __bf16* __restrict__ attnA) {
  __shared__ __bf16 ldsP[8][16 * 40];                  // per-wave P staging (16x32, pad 40)
  const int tid = threadIdx.x, lane = tid & 31, wid = tid >> 5;
  const int hi = lane >> 4, ln = lane & 15;
  const int bh = blockIdx.x >> 4;                      // b*16 + h
  const int qc = blockIdx.x & 15;
  const int b = bh >> 4, h = bh & 15;
  const int q0 = qc * 128 + wid * 16;                  // this wave's first q row

  const __bf16* qrow = Qh + ((size_t)bh * T_SEQ + (q0 + ln)) * HDIM;
  v16bf qf[4];
  #pragma unroll
  for (int dd = 0; dd < 4; ++dd) qf[dd] = load_a(qrow + dd * 32, hi);

  v8f o[8];
  #pragma unroll
  for (int g = 0; g < 8; ++g) o[g] = zero8();
  float mst[8], lst[8];
  #pragma unroll
  for (int r = 0; r < 8; ++r) { mst[r] = -1e30f; lst[r] = 0.0f; }

  __bf16* pw = ldsP[wid];
  const int kend = q0 + 15;
  for (int kt = 0; kt <= kend; kt += 32) {
    v8f s0 = zero8(), s1 = zero8();
    const __bf16* k0p = Kh + ((size_t)bh * T_SEQ + (kt + ln)) * HDIM;
    const __bf16* k1p = k0p + 16 * HDIM;
    #pragma unroll
    for (int dd = 0; dd < 4; ++dd) {                   // contract over hd=128
      v16bf kb0 = load_b(k0p + dd * 32, hi);
      v16bf kb1 = load_b(k1p + dd * 32, hi);
      s0 = wmma_bf16(qf[dd], kb0, s0);
      s1 = wmma_bf16(qf[dd], kb1, s1);
    }
    // online softmax; C-frag row = hi*8+r, col = ln
    #pragma unroll
    for (int r = 0; r < 8; ++r) {
      int q = q0 + hi * 8 + r;
      float a0 = (kt + ln      <= q) ? s0[r] : -1e30f;
      float a1 = (kt + 16 + ln <= q) ? s1[r] : -1e30f;
      float mx = fmaxf(a0, a1);
      #pragma unroll
      for (int msk = 8; msk >= 1; msk >>= 1) mx = fmaxf(mx, __shfl_xor(mx, msk, 16));
      float mnew = fmaxf(mst[r], mx);
      float scale = __expf(mst[r] - mnew);
      float p0 = (a0 > -1e29f) ? __expf(a0 - mnew) : 0.0f;
      float p1 = (a1 > -1e29f) ? __expf(a1 - mnew) : 0.0f;
      float rs = p0 + p1;
      #pragma unroll
      for (int msk = 8; msk >= 1; msk >>= 1) rs += __shfl_xor(rs, msk, 16);
      lst[r] = lst[r] * scale + rs;
      mst[r] = mnew;
      #pragma unroll
      for (int g = 0; g < 8; ++g) o[g][r] *= scale;
      pw[(hi * 8 + r) * 40 + ln]      = (__bf16)p0;
      pw[(hi * 8 + r) * 40 + 16 + ln] = (__bf16)p1;
    }
    asm volatile("s_wait_dscnt 0" ::: "memory");       // P store -> P A-frag load (same wave)
    v16bf pa = load_a(&pw[ln * 40], hi);
    const __bf16* vcol = VhT + ((size_t)bh * HDIM) * T_SEQ + kt;
    #pragma unroll
    for (int g = 0; g < 8; ++g) {
      v16bf vb = load_b(vcol + (size_t)(g * 16 + ln) * T_SEQ, hi);
      o[g] = wmma_bf16(pa, vb, o[g]);
    }
  }
  #pragma unroll
  for (int r = 0; r < 8; ++r) {
    float inv = 1.0f / lst[r];
    int q = q0 + hi * 8 + r;
    __bf16* orow = attnA + ((size_t)(b * T_SEQ + q)) * D_MODEL + h * HDIM + ln;
    #pragma unroll
    for (int g = 0; g < 8; ++g) orow[g * 16] = (__bf16)(o[g][r] * inv);
  }
}

// ---------------- host launcher ----------------
extern "C" void kernel_launch(void* const* d_in, const int* in_sizes, int n_in,
                              void* d_out, int out_size, void* d_ws, size_t ws_size,
                              hipStream_t stream) {
  (void)n_in; (void)out_size; (void)ws_size;
  const float* x          = (const float*)d_in[0];
  const float* gate_w     = (const float*)d_in[3];
  const float* gate_b     = (const float*)d_in[4];
  const float* phi_res    = (const float*)d_in[5];
  const float* ln1_w      = (const float*)d_in[6];
  const float* ln2_w      = (const float*)d_in[7];
  const float* in_proj_w  = (const float*)d_in[8];
  const float* in_proj_b  = (const float*)d_in[9];
  const float* out_proj_w = (const float*)d_in[10];
  const float* out_proj_b = (const float*)d_in[11];
  const float* mlp_w1     = (const float*)d_in[12];
  const float* mlp_b1     = (const float*)d_in[13];
  const float* mlp_w2     = (const float*)d_in[14];
  const float* mlp_b2     = (const float*)d_in[15];
  float* out = (float*)d_out;

  const int B = in_sizes[0] / (T_SEQ * D_MODEL);
  const int M = B * T_SEQ;

  char* ws = (char*)d_ws;
  size_t off = 0;
  auto take = [&](size_t bytes) -> char* {
    char* p = ws + off; off += (bytes + 255) & ~(size_t)255; return p;
  };
  float*  x_state = (float*) take((size_t)B * D_MODEL * 4);
  float*  g_res   = (float*) take((size_t)B * NSTR * 4);
  float*  H_res   = (float*) take(64 * 4);
  __bf16* w_qkv   = (__bf16*)take((size_t)3 * D_MODEL * D_MODEL * 2);
  __bf16* w_out   = (__bf16*)take((size_t)D_MODEL * D_MODEL * 2);
  __bf16* w_m1    = (__bf16*)take((size_t)4 * D_MODEL * D_MODEL * 2);
  __bf16* w_m2    = (__bf16*)take((size_t)4 * D_MODEL * D_MODEL * 2);
  __bf16* normA   = (__bf16*)take((size_t)M * D_MODEL * 2);
  __bf16* Qh      = (__bf16*)take((size_t)M * D_MODEL * 2);
  __bf16* Kh      = (__bf16*)take((size_t)M * D_MODEL * 2);
  __bf16* VhT     = (__bf16*)take((size_t)M * D_MODEL * 2);
  __bf16* attnA   = (__bf16*)take((size_t)M * D_MODEL * 2);
  float*  x2      = (float*) take((size_t)M * D_MODEL * 4);
  __bf16* hbuf    = (__bf16*)take((size_t)M * 4 * D_MODEL * 2);

  // --- mHC gated stream-mix path (writes mhc_out into d_out) ---
  mean_kernel<<<B * (D_MODEL / 256), 256, 0, stream>>>(x, x_state);
  gate_kernel<<<B * NSTR, 256, 0, stream>>>(x_state, gate_w, gate_b, g_res);
  sinkhorn_kernel<<<1, 64, 0, stream>>>(phi_res, H_res);
  mhc_kernel<<<B * T_SEQ, 256, 0, stream>>>(x, H_res, g_res, out);

  // --- weight precision: fp32 -> bf16 ---
  f32_to_bf16_kernel<<<(3 * D_MODEL * D_MODEL) / 256, 256, 0, stream>>>(in_proj_w, w_qkv);
  f32_to_bf16_kernel<<<(D_MODEL * D_MODEL) / 256, 256, 0, stream>>>(out_proj_w, w_out);
  f32_to_bf16_kernel<<<(4 * D_MODEL * D_MODEL) / 256, 256, 0, stream>>>(mlp_w1, w_m1);
  f32_to_bf16_kernel<<<(4 * D_MODEL * D_MODEL) / 256, 256, 0, stream>>>(mlp_w2, w_m2);

  // --- attention path ---
  rmsnorm_bf16_kernel<<<M, 256, 0, stream>>>(x, ln1_w, normA);
  gemm_wmma_kernel<0><<<dim3(3 * D_MODEL / 128, M / 128), 256, 0, stream>>>(
      normA, w_qkv, in_proj_b, M, 3 * D_MODEL, D_MODEL,
      nullptr, nullptr, nullptr, nullptr, Qh, Kh, VhT);
  attention_kernel<<<B * NHEADS * (T_SEQ / 128), 256, 0, stream>>>(Qh, Kh, VhT, attnA);
  gemm_wmma_kernel<1><<<dim3(D_MODEL / 128, M / 128), 256, 0, stream>>>(
      attnA, w_out, out_proj_b, M, D_MODEL, D_MODEL,
      x2, nullptr, x, nullptr, nullptr, nullptr, nullptr);

  // --- MLP path + final fuse (x2 + ffn + mhc) ---
  rmsnorm_bf16_kernel<<<M, 256, 0, stream>>>(x2, ln2_w, normA);
  gemm_wmma_kernel<2><<<dim3(4 * D_MODEL / 128, M / 128), 256, 0, stream>>>(
      normA, w_m1, mlp_b1, M, 4 * D_MODEL, D_MODEL,
      nullptr, hbuf, nullptr, nullptr, nullptr, nullptr, nullptr);
  gemm_wmma_kernel<3><<<dim3(D_MODEL / 128, M / 128), 256, 0, stream>>>(
      hbuf, w_m2, mlp_b2, M, D_MODEL, 4 * D_MODEL,
      out, nullptr, x2, out, nullptr, nullptr, nullptr);
}
